// StackedTransEncoder_24816321036434
// MI455X (gfx1250) — matrix operand order repaired and checked by reading, not compiled, not run on hardware
//
#include <hip/hip_runtime.h>
#include <stdint.h>

// ---------------- problem constants ----------------
#define B_   128
#define L_   256
#define E_   512
#define H_   512
#define G3H  1536                 // 3*H
#define NWG  16                   // workgroups per direction
#define NTHR 256                  // 8 waves (wave32)
#define LDP  520                  // padded LDS row stride (bf16 elems) -> 1040B, rotates banks
#define FRAGS_C0  3072            // (1024/32) * (1536/16)
#define FRAGS_C12 1536            // (512/32)  * (1536/16)
#define FRAGS_DIR (FRAGS_C0 + 2*FRAGS_C12)   // 6144

// ---------------- workspace layout (bytes) ----------------
static constexpr size_t OFF_PACKW = 0;
static constexpr size_t SZ_PACKW  = 2ull * FRAGS_DIR * 1024;         // 12,582,912
static constexpr size_t OFF_BIAS  = OFF_PACKW + SZ_PACKW;
static constexpr size_t SZ_BIAS   = 2ull * 3 * G3H * 4;              // 36,864
static constexpr size_t OFF_HBUF  = OFF_BIAS + SZ_BIAS;              // bf16 [2][2][128][512]
static constexpr size_t SZ_HBUF   = 2ull * 2 * B_ * H_ * 2;          // 524,288
static constexpr size_t OFF_SYNC  = OFF_HBUF + SZ_HBUF;              // uint [2] (stride 32)
static constexpr size_t SZ_SYNC   = 256;

// ---------------- WMMA / TDM types ----------------
typedef __bf16 v16bf __attribute__((ext_vector_type(16)));
typedef float  v8f   __attribute__((ext_vector_type(8)));
typedef unsigned int u32x4 __attribute__((ext_vector_type(4)));
typedef int          i32x8 __attribute__((ext_vector_type(8)));
typedef int          i32x4 __attribute__((ext_vector_type(4)));

union Frag16 { uint4 u[2]; v16bf v; };
union U4BF8  { unsigned short s[8]; uint4 u; };

#define WMMA(Af, Bf, Cf) \
  __builtin_amdgcn_wmma_f32_16x16x32_bf16(false, (Af), false, (Bf), (short)0, (Cf), false, false)

#if __has_builtin(__builtin_amdgcn_tensor_load_to_lds) && __has_builtin(__builtin_amdgcn_s_wait_tensorcnt)
#define USE_TDM 1
#else
#define USE_TDM 0
#endif

__device__ __forceinline__ unsigned short bf16bits(float f) {
  union { __bf16 b; unsigned short s; } u; u.b = (__bf16)f; return u.s;
}
__device__ __forceinline__ float bf16val(unsigned short s) {
  union { unsigned short s; __bf16 b; } u; u.s = s; return (float)u.b;
}
__device__ __forceinline__ float sigmoidf_(float x) {
  return 1.0f / (1.0f + __expf(-x));
}

// =====================================================================
// Pack fp32 weights -> bf16 WMMA B-fragment layout + gather biases.
// B fragment (32x16 bf16): n = nt*16 + (lane&15), k = kt*32 + j + ((lane>>4)<<4).
// Cell 0 K = [x(512) | h(512)] rows: Wx then Uh.
// =====================================================================
__global__ void pack_kernel(const float* __restrict__ f_Wx, const float* __restrict__ f_Uh,
                            const float* __restrict__ f_b,  const float* __restrict__ b_Wx,
                            const float* __restrict__ b_Uh, const float* __restrict__ b_b,
                            const float* __restrict__ f_tU, const float* __restrict__ f_tb,
                            const float* __restrict__ b_tU, const float* __restrict__ b_tb,
                            unsigned short* __restrict__ packW, float* __restrict__ biasbuf)
{
  long long tid = (long long)blockIdx.x * blockDim.x + threadIdx.x;
  const long long NW = 2ll * FRAGS_DIR * 512;
  if (tid < NW) {
    int j    = (int)(tid & 15);
    int lane = (int)((tid >> 4) & 31);
    long long fg = tid >> 9;
    int d = (int)(fg / FRAGS_DIR);
    int f = (int)(fg % FRAGS_DIR);
    int cell, fi;
    if      (f < FRAGS_C0)             { cell = 0; fi = f; }
    else if (f < FRAGS_C0 + FRAGS_C12) { cell = 1; fi = f - FRAGS_C0; }
    else                               { cell = 2; fi = f - FRAGS_C0 - FRAGS_C12; }
    int kt = fi / 96, nt = fi % 96;
    int n  = nt * 16 + (lane & 15);
    int kk = kt * 32 + j + ((lane >> 4) << 4);
    const float* Wx = d ? b_Wx : f_Wx;
    const float* Uh = d ? b_Uh : f_Uh;
    const float* tU = d ? b_tU : f_tU;
    float v;
    if (cell == 0) v = (kk < E_) ? Wx[(long long)kk * G3H + n]
                                 : Uh[(long long)(kk - E_) * G3H + n];
    else           v = tU[((long long)(cell - 1) * H_ + kk) * G3H + n];
    packW[tid] = bf16bits(v);
  } else {
    long long bid = tid - NW;
    if (bid < 2ll * 3 * G3H) {
      int d = (int)(bid / (3 * G3H));
      int r = (int)(bid % (3 * G3H));
      int cell = r / G3H, n = r % G3H;
      float v = (cell == 0) ? (d ? b_b[n] : f_b[n])
                            : (d ? b_tb[(cell - 1) * G3H + n] : f_tb[(cell - 1) * G3H + n]);
      biasbuf[(d * 3 + cell) * G3H + n] = v;
    }
  }
}

// =====================================================================
// One 16-iteration K sweep: A frags from LDS, B frags from packed global,
// accumulating (r0,r1,z0,z1,P0,P1) where P is the h-tilde section target.
// =====================================================================
__device__ __forceinline__ void gemm16(const unsigned short* __restrict__ Asrc,  // lds + row*LDP + akoff
                                       const unsigned short* __restrict__ Bbase, // packD + fragbase*512 + lane*16
                                       int wg,
                                       v8f& aR0, v8f& aR1, v8f& aZ0, v8f& aZ1,
                                       v8f& aP0, v8f& aP1)
{
  for (int kt = 0; kt < 16; ++kt) {
    Frag16 a;
    const unsigned short* Ab = Asrc + kt * 32;
    a.u[0] = *(const uint4*)(Ab);        // K {base..base+7}
    a.u[1] = *(const uint4*)(Ab + 8);    // K {base+8..base+15}
    const unsigned short* Bk = Bbase + (size_t)kt * 96 * 512;
    if (kt + 1 < 16) __builtin_prefetch(Bk + 96 * 512, 0, 1);
#pragma unroll
    for (int nt = 0; nt < 6; ++nt) {
      const int ntg = (nt >> 1) * 32 + (wg << 1) + (nt & 1);   // global N-tile (of 96)
      const unsigned short* Bp = Bk + (size_t)ntg * 512;
      Frag16 b;
      b.u[0] = *(const uint4*)(Bp);
      b.u[1] = *(const uint4*)(Bp + 8);
      if      (nt == 0) aR0 = WMMA(a.v, b.v, aR0);
      else if (nt == 1) aR1 = WMMA(a.v, b.v, aR1);
      else if (nt == 2) aZ0 = WMMA(a.v, b.v, aZ0);
      else if (nt == 3) aZ1 = WMMA(a.v, b.v, aZ1);
      else if (nt == 4) aP0 = WMMA(a.v, b.v, aP0);
      else              aP1 = WMMA(a.v, b.v, aP1);
    }
  }
}

// =====================================================================
// Persistent scan: 2 directions x 16 WGs. Each WG owns 32 hidden columns.
// Per (t, cell): TDM-stage h (and manually cvt-stage x_t for cell 0) in LDS,
// WMMA-GEMM the owned 96 gate columns, gate math, write bf16 h slice,
// device sync via release/acquire atomic counter.
// =====================================================================
__global__ void __launch_bounds__(NTHR, 1)
gru_scan_kernel(const float* __restrict__ xs, const float* __restrict__ xmask,
                const unsigned short* __restrict__ packW,
                const float* __restrict__ biasbuf,
                unsigned short* __restrict__ hb,     // [2][2][B][H] bf16
                unsigned int* __restrict__ sync,     // [2] (stride 32 uints)
                float* __restrict__ out)             // [B][L][2H]
{
  extern __shared__ __align__(16) unsigned short smem[];
  unsigned short* xlds = smem;                  // [B_][LDP] bf16 (x_t)
  unsigned short* hlds = smem + B_ * LDP;       // [B_][LDP] bf16 (h)

  const int d    = blockIdx.x / NWG;
  const int wg   = blockIdx.x % NWG;
  const int tid  = threadIdx.x;
  const int wave = tid >> 5;
  const int lane = tid & 31;
  const int c0   = wg * 32;                     // owned hidden-column base
  const int m0   = wave * 16;                   // owned batch-row tile
  const int arow = m0 + (lane & 15);
  const int akoff = (lane & 16) ? 16 : 0;       // A-frag K sub-offset: lanes 16-31 hold k 16..31 (contiguous)
  const int ncol  = lane & 15;
  const int rowOff = (lane & 16) ? 8 : 0;

  unsigned int* cnt = sync + d * 32;
  const float* biasd = biasbuf + d * 3 * G3H;
  const unsigned short* packD = packW + (size_t)d * FRAGS_DIR * 512;
  const unsigned short* AsrcX = xlds + arow * LDP + akoff;
  const unsigned short* AsrcH = hlds + arow * LDP + akoff;

  unsigned int phase = 0;

  for (int step = 0; step < L_; ++step) {
    const int t = d ? (L_ - 1 - step) : step;

    // mask per owned row, constant over the 3 cells of this timestep
    float mvals[8];
#pragma unroll
    for (int e = 0; e < 8; ++e)
      mvals[e] = xmask[(size_t)(m0 + rowOff + e) * L_ + t];

    for (int cell = 0; cell < 3; ++cell) {
      // ---- wait until all WGs of this direction finished previous phase ----
      if (tid == 0) {
        const unsigned int target = phase * NWG;
        while (__hip_atomic_load(cnt, __ATOMIC_ACQUIRE, __HIP_MEMORY_SCOPE_AGENT) < target)
          __builtin_amdgcn_s_sleep(2);
      }
      __syncthreads();

      const unsigned short* hsrc = hb + (((size_t)d * 2 + (phase & 1u)) * B_ * H_);
      unsigned short*       hdst = hb + (((size_t)d * 2 + ((phase + 1u) & 1u)) * B_ * H_);

      // ---- stage h into LDS ----
#if USE_TDM
      // Tensor Data Mover: 2D bf16 tile [128 x 512], row stride 512 elems,
      // LDS pad insertion: 4 DWORDs (16B) after every 256 DWORDs (one row)
      // -> reproduces the LDP=520 padded layout in-flight.
      if (wave == 0) {
        const unsigned long long ga = (unsigned long long)(uintptr_t)hsrc;
        u32x4 g0;
        g0[0] = 1u;                                     // count=1 (valid, user D#)
        g0[1] = (unsigned)(B_ * LDP * 2);               // lds_addr (bytes): hlds region
        g0[2] = (unsigned)(ga & 0xFFFFFFFFu);           // global_addr[31:0]
        g0[3] = (unsigned)((ga >> 32) & 0x01FFFFFFu)    // global_addr[56:32]
              | (2u << 30);                             // type=2 ("image")
        i32x8 g1;
        g1[0] = (int)((1u << 16)      // data_size = 1 -> 2 bytes
                    | (1u << 20)      // pad_enable
                    | (7u << 22)      // pad_interval: 256 DWORDs
                    | (3u << 25));    // pad_amount:   4 DWORDs
        g1[1] = (int)((H_ & 0xFFFFu) << 16);                  // tensor_dim0[15:0]=512
        g1[2] = (int)(((unsigned)H_ >> 16) | ((B_ & 0xFFFFu) << 16)); // dim0 hi | tensor_dim1=128
        g1[3] = (int)(((unsigned)B_ >> 16) | ((H_ & 0xFFFFu) << 16)); // dim1 hi | tile_dim0=512
        g1[4] = (int)(B_ & 0xFFFFu);                          // tile_dim1=128, tile_dim2=0
        g1[5] = (int)H_;                                      // tensor_dim0_stride=512
        g1[6] = 0; g1[7] = 0;
        i32x4 gz4 = {0, 0, 0, 0};
        i32x8 gz8 = {0, 0, 0, 0, 0, 0, 0, 0};
        __builtin_amdgcn_tensor_load_to_lds(g0, g1, gz4, gz4, gz8, 0);
        __builtin_amdgcn_s_wait_tensorcnt(0);
      }
#else
      for (int c = tid; c < (B_ * H_) / 8; c += NTHR) {
        int row = c >> 6;
        int e0  = (c & 63) << 3;
        *(uint4*)(hlds + row * LDP + e0) = *(const uint4*)(hsrc + row * H_ + e0);
      }
#endif
      // ---- stage x_t (fp32 -> bf16) for the LGRU cell ----
      if (cell == 0) {
        for (int c = tid; c < (B_ * E_) / 8; c += NTHR) {
          int row = c >> 6;
          int e0  = (c & 63) << 3;
          const float* xp = xs + ((size_t)row * L_ + t) * E_ + e0;
          float4 a = *(const float4*)xp;
          float4 b = *(const float4*)(xp + 4);
          U4BF8 o;
          o.s[0] = bf16bits(a.x); o.s[1] = bf16bits(a.y);
          o.s[2] = bf16bits(a.z); o.s[3] = bf16bits(a.w);
          o.s[4] = bf16bits(b.x); o.s[5] = bf16bits(b.y);
          o.s[6] = bf16bits(b.z); o.s[7] = bf16bits(b.w);
          *(uint4*)(xlds + row * LDP + e0) = o.u;
        }
      }
      __syncthreads();

      // ---- WMMA GEMM: 1 M-tile x 6 N-tiles per wave ----
      v8f aR0 = {}, aR1 = {}, aZ0 = {}, aZ1 = {};
      v8f aX0 = {}, aX1 = {}, aH0 = {}, aH1 = {};
      if (cell == 0) {
        // K = [x(512) | h(512)]: x-part -> aX (gx for h-tilde), h-part -> aH
        gemm16(AsrcX, packD + lane * 16,                          wg, aR0, aR1, aZ0, aZ1, aX0, aX1);
        gemm16(AsrcH, packD + (size_t)(16 * 96) * 512 + lane * 16, wg, aR0, aR1, aZ0, aZ1, aH0, aH1);
      } else if (cell == 1) {
        gemm16(AsrcH, packD + (size_t)FRAGS_C0 * 512 + lane * 16,  wg, aR0, aR1, aZ0, aZ1, aH0, aH1);
      } else {
        gemm16(AsrcH, packD + (size_t)(FRAGS_C0 + FRAGS_C12) * 512 + lane * 16,
               wg, aR0, aR1, aZ0, aZ1, aH0, aH1);
      }

      // ---- gate math (C/D layout: col = n0+(lane&15), rows m0+{0..7|8..15}) ----
      const float* bs = biasd + cell * G3H;
#pragma unroll
      for (int p = 0; p < 2; ++p) {
        const v8f gr = p ? aR1 : aR0;
        const v8f gz = p ? aZ1 : aZ0;
        const v8f gx = p ? aX1 : aX0;
        const v8f gh = p ? aH1 : aH0;
        const int col = c0 + p * 16 + ncol;
        const float bR = bs[col];
        const float bZ = bs[H_ + col];
        const float bH = bs[2 * H_ + col];
#pragma unroll
        for (int e = 0; e < 8; ++e) {
          const int row = m0 + rowOff + e;
          const float hp = bf16val(hlds[row * LDP + col]);
          const float r  = sigmoidf_(gr[e] + bR);
          const float z  = sigmoidf_(gz[e] + bZ);
          const float ht = tanhf(gx[e] + bH + r * gh[e]);   // gx=0 for TGRU cells
          const float hn = (1.0f - z) * ht + z * hp;
          const float m  = mvals[e];
          const float hnew = m * hn + (1.0f - m) * hp;
          hdst[(size_t)row * H_ + col] = bf16bits(hnew);
          if (cell == 2)
            out[((size_t)row * L_ + t) * (2 * H_) + d * H_ + col] = hnew * m;
        }
      }

      // ---- publish + arrive ----
      __threadfence();
      __syncthreads();
      if (tid == 0)
        __hip_atomic_fetch_add(cnt, 1u, __ATOMIC_RELEASE, __HIP_MEMORY_SCOPE_AGENT);
      ++phase;
    }
  }
}

// =====================================================================
extern "C" void kernel_launch(void* const* d_in, const int* in_sizes, int n_in,
                              void* d_out, int out_size, void* d_ws, size_t ws_size,
                              hipStream_t stream)
{
  (void)in_sizes; (void)n_in; (void)out_size; (void)ws_size;
  const float* xs    = (const float*)d_in[0];
  const float* xmask = (const float*)d_in[1];
  const float* f_Wx  = (const float*)d_in[2];
  const float* f_Uh  = (const float*)d_in[3];
  const float* f_b   = (const float*)d_in[4];
  const float* b_Wx  = (const float*)d_in[5];
  const float* b_Uh  = (const float*)d_in[6];
  const float* b_b   = (const float*)d_in[7];
  const float* f_tU  = (const float*)d_in[8];
  const float* f_tb  = (const float*)d_in[9];
  const float* b_tU  = (const float*)d_in[10];
  const float* b_tb  = (const float*)d_in[11];

  char* ws = (char*)d_ws;
  unsigned short* packW  = (unsigned short*)(ws + OFF_PACKW);
  float*          biasbf = (float*)(ws + OFF_BIAS);
  unsigned short* hb     = (unsigned short*)(ws + OFF_HBUF);
  unsigned int*   syncp  = (unsigned int*)(ws + OFF_SYNC);

  // zero h state (h0 = 0) and sync counters every call (deterministic)
  (void)hipMemsetAsync(ws + OFF_HBUF, 0, SZ_HBUF + SZ_SYNC, stream);

  const long long npack = 2ll * FRAGS_DIR * 512 + 2ll * 3 * G3H;
  const int pblocks = (int)((npack + 255) / 256);
  pack_kernel<<<pblocks, 256, 0, stream>>>(f_Wx, f_Uh, f_b, b_Wx, b_Uh, b_b,
                                           f_tU, f_tb, b_tU, b_tb, packW, biasbf);

  const int smem = 2 * B_ * LDP * (int)sizeof(unsigned short);  // 266,240 B (<320KB WGP LDS)
  gru_scan_kernel<<<2 * NWG, NTHR, smem, stream>>>(xs, xmask, packW, biasbf,
                                                   hb, syncp, (float*)d_out);
}
